// CrossAttention_89069031784468
// MI455X (gfx1250) — compile-verified, matrix operand order checked by reading, not run on hardware
//
#include <hip/hip_runtime.h>

// ---------------------------------------------------------------------------
// CDNA5 (gfx1250) cross-attention:
//   - bf16 WMMA projections (fused K/V GEMM sharing A-fragment loads)
//   - flash attention with double-buffered GLOBAL_LOAD_ASYNC_TO_LDS_B128
//     staging of block-shared K/V chunks (ASYNCcnt pipelined against WMMA).
// ---------------------------------------------------------------------------

typedef __attribute__((ext_vector_type(16))) __bf16 bf16x16;
typedef __attribute__((ext_vector_type(8)))  float  f32x8;

union AFrag { bf16x16 v; __bf16 e[16]; unsigned int u[8]; };
union CFrag { f32x8  v; float  e[8]; };

// K-offset held by VGPR j for a 16-bit 16x32 A/B fragment (ISA 7.12.2).
__device__ __forceinline__ int koff(int j, int half) {
  return ((j >> 2) << 4) + (half << 3) + ((j & 3) << 1);
}

// Low 32 bits of a generic shared-memory address = LDS byte offset (ISA §10.2).
__device__ __forceinline__ unsigned lds_off(const void* p) {
  return (unsigned)(unsigned long long)p;
}

// Async DMA of 16 bytes/lane global -> LDS, tracked by ASYNCcnt (§15.18 op 98).
__device__ __forceinline__ void async_copy_b128(unsigned lds_byte, const void* g) {
  asm volatile("global_load_async_to_lds_b128 %0, %1, off"
               :: "v"(lds_byte), "v"(g) : "memory");
}

// ---------------------------------------------------------------------------
// Q projection: Y[m,n] = sum_d X[m,d] * W[n,d] + b[n], Y stored bf16.
// Each wave computes a 16x64 strip (4 accumulators) reusing its A fragment 4x.
// ---------------------------------------------------------------------------
__global__ __launch_bounds__(256)
void ca_proj_bf16(const float* __restrict__ Xin, const float* __restrict__ W,
                  const float* __restrict__ bias, __bf16* __restrict__ Yout,
                  int M, int Din, int Dout)
{
  const int lane = threadIdx.x & 31;
  const int wave = threadIdx.x >> 5;
  const int half = lane >> 4;
  const int nlan = lane & 15;
  const int m0 = blockIdx.x << 4;
  const int n0 = (blockIdx.y << 9) + (wave << 6);   // 512 cols per block

  const float* arow = Xin + (size_t)(m0 + nlan) * Din;
  const float* brow[4];
  float bn[4];
#pragma unroll
  for (int s = 0; s < 4; ++s) {
    brow[s] = W + (size_t)(n0 + s * 16 + nlan) * Din;
    bn[s]   = bias[n0 + s * 16 + nlan];
  }

  CFrag acc[4];
#pragma unroll
  for (int s = 0; s < 4; ++s)
#pragma unroll
    for (int r = 0; r < 8; ++r) acc[s].e[r] = 0.f;

  for (int d0 = 0; d0 < Din; d0 += 32) {
    AFrag a;
#pragma unroll
    for (int j = 0; j < 8; ++j) {
      float2 fa = *(const float2*)(arow + d0 + koff(j, half));
      a.e[2*j] = (__bf16)fa.x; a.e[2*j+1] = (__bf16)fa.y;
    }
#pragma unroll
    for (int s = 0; s < 4; ++s) {
      AFrag b;
#pragma unroll
      for (int j = 0; j < 8; ++j) {
        float2 fb = *(const float2*)(brow[s] + d0 + koff(j, half));
        b.e[2*j] = (__bf16)fb.x; b.e[2*j+1] = (__bf16)fb.y;
      }
      acc[s].v = __builtin_amdgcn_wmma_f32_16x16x32_bf16(false, a.v, false, b.v,
                                                         (short)0, acc[s].v,
                                                         false, false);
    }
  }

#pragma unroll
  for (int s = 0; s < 4; ++s)
#pragma unroll
    for (int r = 0; r < 8; ++r) {
      const int m = (half << 3) + r;
      Yout[(size_t)(m0 + m) * Dout + n0 + s * 16 + nlan] =
          (__bf16)(acc[s].e[r] + bn[s]);
    }
}

// ---------------------------------------------------------------------------
// Fused K+V projection: both GEMMs consume the same A (context) rows, so each
// A fragment is loaded/converted once and feeds two WMMA accumulator chains.
// ---------------------------------------------------------------------------
__global__ __launch_bounds__(256)
void ca_proj_kv_bf16(const float* __restrict__ Ctx,
                     const float* __restrict__ Wk, const float* __restrict__ bk,
                     const float* __restrict__ Wv, const float* __restrict__ bv,
                     __bf16* __restrict__ Kout, __bf16* __restrict__ Vout,
                     int M, int Din, int Dout)
{
  const int lane = threadIdx.x & 31;
  const int wave = threadIdx.x >> 5;
  const int half = lane >> 4;
  const int nlan = lane & 15;
  const int m0 = blockIdx.x << 4;
  const int n0 = (blockIdx.y << 9) + (wave << 6);

  const float* arow = Ctx + (size_t)(m0 + nlan) * Din;
  const float* krow[4];
  const float* vrow[4];
  float kbn[4], vbn[4];
#pragma unroll
  for (int s = 0; s < 4; ++s) {
    const int n = n0 + s * 16 + nlan;
    krow[s] = Wk + (size_t)n * Din;  kbn[s] = bk[n];
    vrow[s] = Wv + (size_t)n * Din;  vbn[s] = bv[n];
  }

  CFrag ka[4], va[4];
#pragma unroll
  for (int s = 0; s < 4; ++s)
#pragma unroll
    for (int r = 0; r < 8; ++r) { ka[s].e[r] = 0.f; va[s].e[r] = 0.f; }

  for (int d0 = 0; d0 < Din; d0 += 32) {
    AFrag a;
#pragma unroll
    for (int j = 0; j < 8; ++j) {
      float2 fa = *(const float2*)(arow + d0 + koff(j, half));
      a.e[2*j] = (__bf16)fa.x; a.e[2*j+1] = (__bf16)fa.y;
    }
#pragma unroll
    for (int s = 0; s < 4; ++s) {
      AFrag b;
#pragma unroll
      for (int j = 0; j < 8; ++j) {
        float2 fb = *(const float2*)(krow[s] + d0 + koff(j, half));
        b.e[2*j] = (__bf16)fb.x; b.e[2*j+1] = (__bf16)fb.y;
      }
      ka[s].v = __builtin_amdgcn_wmma_f32_16x16x32_bf16(false, a.v, false, b.v,
                                                        (short)0, ka[s].v,
                                                        false, false);
#pragma unroll
      for (int j = 0; j < 8; ++j) {
        float2 fb = *(const float2*)(vrow[s] + d0 + koff(j, half));
        b.e[2*j] = (__bf16)fb.x; b.e[2*j+1] = (__bf16)fb.y;
      }
      va[s].v = __builtin_amdgcn_wmma_f32_16x16x32_bf16(false, a.v, false, b.v,
                                                        (short)0, va[s].v,
                                                        false, false);
    }
  }

#pragma unroll
  for (int s = 0; s < 4; ++s)
#pragma unroll
    for (int r = 0; r < 8; ++r) {
      const int m = (half << 3) + r;
      const size_t idx = (size_t)(m0 + m) * Dout + n0 + s * 16 + nlan;
      Kout[idx] = (__bf16)(ka[s].e[r] + kbn[s]);
      Vout[idx] = (__bf16)(va[s].e[r] + vbn[s]);
    }
}

// ---------------------------------------------------------------------------
// Flash attention. All 8 waves of a block share (batch, head); the 32x64 K/V
// chunks are staged into double-buffered LDS via async-to-LDS DMA so the copy
// of chunk i+1 overlaps the WMMA/softmax work on chunk i.
// ---------------------------------------------------------------------------
namespace {
constexpr int AD  = 1024;   // model dim
constexpr int AH  = 16;     // heads
constexpr int AHD = 64;     // head dim
constexpr int AX  = 1024;   // query length
constexpr int AT  = 4096;   // context length
}

__global__ __launch_bounds__(256)
void ca_flash_attn(const __bf16* __restrict__ Q, const __bf16* __restrict__ K,
                   const __bf16* __restrict__ V, float* __restrict__ Out)
{
  __shared__ __bf16 sK[2][32][64];       // 2 x 4 KB double-buffered K chunk
  __shared__ __bf16 sV[2][32][64];       // 2 x 4 KB double-buffered V chunk
  __shared__ __bf16 sP[8][16 * 32];      // per-wave P staging (8 KB)

  const int tid  = threadIdx.x;
  const int lane = tid & 31;
  const int wave = tid >> 5;
  const int half = lane >> 4;
  const int nlan = lane & 15;

  const int g0 = blockIdx.x * 8;         // waves in a block share (nb, h)
  const int xt = (g0 & 63) + wave;
  const int h  = (g0 >> 6) & (AH - 1);
  const int nb = g0 >> 10;
  const int x0 = xt << 4;

  // Per-thread staging slot: 256 threads x 16 B == one 32x64 bf16 tile.
  const int crow = tid >> 3;             // 0..31  (t within chunk)
  const int ccol = (tid & 7) * 8;        // 0..56  (bf16 elems)
  const unsigned ldsK[2] = { lds_off(&sK[0][0][0]) + (unsigned)tid * 16,
                             lds_off(&sK[1][0][0]) + (unsigned)tid * 16 };
  const unsigned ldsV[2] = { lds_off(&sV[0][0][0]) + (unsigned)tid * 16,
                             lds_off(&sV[1][0][0]) + (unsigned)tid * 16 };

  // Q fragments: 16x64 = two 16x32 bf16 A fragments, loaded once.
  AFrag qa[2];
  {
    const __bf16* qrow = Q + (size_t)(nb * AX + x0 + nlan) * AD + h * AHD;
#pragma unroll
    for (int f = 0; f < 2; ++f)
#pragma unroll
      for (int j = 0; j < 8; ++j)
        qa[f].u[j] = *(const unsigned int*)(qrow + f * 32 + koff(j, half));
  }

  float mr[8], lr[8];
  CFrag o[4];
#pragma unroll
  for (int r = 0; r < 8; ++r) { mr[r] = -3.0e38f; lr[r] = 0.f; }
#pragma unroll
  for (int dt = 0; dt < 4; ++dt)
#pragma unroll
    for (int r = 0; r < 8; ++r) o[dt].e[r] = 0.f;

  const __bf16* kbase = K + (size_t)nb * AT * AD + h * AHD;
  const __bf16* vbase = V + (size_t)nb * AT * AD + h * AHD;

  // Prime the pipeline: chunk 0 -> buffer 0.
  async_copy_b128(ldsK[0], kbase + (size_t)crow * AD + ccol);
  async_copy_b128(ldsV[0], vbase + (size_t)crow * AD + ccol);

  for (int t0 = 0, it = 0; t0 < AT; t0 += 32, ++it) {
    const int b = it & 1;
    if (t0 + 32 < AT) {
      // Issue chunk it+1 into the other buffer; it flies under this chunk's
      // compute. Buffer b^1 was last read in iteration it-1 (barrier-protected).
      async_copy_b128(ldsK[b ^ 1], kbase + (size_t)(t0 + 32 + crow) * AD + ccol);
      async_copy_b128(ldsV[b ^ 1], vbase + (size_t)(t0 + 32 + crow) * AD + ccol);
      __builtin_prefetch(kbase + (size_t)(t0 + 64 + crow) * AD + ccol, 0, 1);
      __builtin_prefetch(vbase + (size_t)(t0 + 64 + crow) * AD + ccol, 0, 1);
      // ASYNCcnt completes in order: <=2 outstanding means chunk `it` landed.
      asm volatile("s_wait_asynccnt 2" ::: "memory");
    } else {
      asm volatile("s_wait_asynccnt 0" ::: "memory");
    }
    __syncthreads();                     // chunk `it` visible to all waves

    // ---- S = Q @ K^T for two 16-column tiles (K-loop over head dim 64) ----
    CFrag s0, s1;
#pragma unroll
    for (int r = 0; r < 8; ++r) { s0.e[r] = 0.f; s1.e[r] = 0.f; }

#pragma unroll
    for (int f = 0; f < 2; ++f) {
      AFrag b0, b1;
#pragma unroll
      for (int j = 0; j < 8; ++j) {
        const int k = f * 32 + koff(j, half);
        b0.u[j] = *(const unsigned int*)&sK[b][nlan][k];
        b1.u[j] = *(const unsigned int*)&sK[b][16 + nlan][k];
      }
      s0.v = __builtin_amdgcn_wmma_f32_16x16x32_bf16(false, qa[f].v, false, b0.v,
                                                     (short)0, s0.v, false, false);
      s1.v = __builtin_amdgcn_wmma_f32_16x16x32_bf16(false, qa[f].v, false, b1.v,
                                                     (short)0, s1.v, false, false);
    }

    // ---- online softmax (scale 1/sqrt(64) = 0.125) ----
    float p0[8], p1[8], sc[8];
#pragma unroll
    for (int r = 0; r < 8; ++r) {
      const float a = s0.e[r] * 0.125f;
      const float bb = s1.e[r] * 0.125f;
      float c = fmaxf(a, bb);
      c = fmaxf(c, __shfl_xor(c, 1));
      c = fmaxf(c, __shfl_xor(c, 2));
      c = fmaxf(c, __shfl_xor(c, 4));
      c = fmaxf(c, __shfl_xor(c, 8));
      const float mn = fmaxf(mr[r], c);
      sc[r] = __expf(mr[r] - mn);
      p0[r] = __expf(a - mn);
      p1[r] = __expf(bb - mn);
      float rs = p0[r] + p1[r];
      rs += __shfl_xor(rs, 1);
      rs += __shfl_xor(rs, 2);
      rs += __shfl_xor(rs, 4);
      rs += __shfl_xor(rs, 8);
      lr[r] = lr[r] * sc[r] + rs;
      mr[r] = mn;
    }
#pragma unroll
    for (int dt = 0; dt < 4; ++dt)
#pragma unroll
      for (int r = 0; r < 8; ++r) o[dt].e[r] *= sc[r];

    // ---- stage P (16x32 bf16) through LDS: C layout -> A layout ----
    __bf16* lp = &sP[wave][0];
#pragma unroll
    for (int r = 0; r < 8; ++r) {
      const int m = (half << 3) + r;
      lp[m * 32 + nlan]      = (__bf16)p0[r];
      lp[m * 32 + 16 + nlan] = (__bf16)p1[r];
    }
    asm volatile("s_wait_dscnt 0" ::: "memory");   // same-wave LDS RAW
    AFrag pa;
#pragma unroll
    for (int j = 0; j < 8; ++j)
      pa.u[j] = *(const unsigned int*)(lp + nlan * 32 + koff(j, half));

    // ---- O += P @ V_chunk (four 16-wide head-dim tiles), V from LDS ----
#pragma unroll
    for (int dt = 0; dt < 4; ++dt) {
      AFrag bv;
#pragma unroll
      for (int j = 0; j < 8; ++j) {
        const int k = koff(j, half);
        bv.e[2*j]   = sV[b][k][dt * 16 + nlan];
        bv.e[2*j+1] = sV[b][k + 1][dt * 16 + nlan];
      }
      o[dt].v = __builtin_amdgcn_wmma_f32_16x16x32_bf16(false, pa.v, false, bv.v,
                                                        (short)0, o[dt].v,
                                                        false, false);
    }
    __syncthreads();   // buffer b fully read before iteration it+1 overwrites it
  }

  // ---- normalize and store fp32 output (N, X, D) with head interleave ----
  float inv[8];
#pragma unroll
  for (int r = 0; r < 8; ++r) inv[r] = 1.0f / lr[r];
#pragma unroll
  for (int dt = 0; dt < 4; ++dt)
#pragma unroll
    for (int r = 0; r < 8; ++r) {
      const int m = (half << 3) + r;
      Out[(size_t)(nb * AX + x0 + m) * AD + h * AHD + dt * 16 + nlan] =
          o[dt].e[r] * inv[r];
    }
}

// ---------------------------------------------------------------------------
// Host launcher
// ---------------------------------------------------------------------------
extern "C" void kernel_launch(void* const* d_in, const int* in_sizes, int n_in,
                              void* d_out, int out_size, void* d_ws, size_t ws_size,
                              hipStream_t stream)
{
  (void)n_in; (void)out_size; (void)ws_size;

  const float* prev = (const float*)d_in[0];  // (N, X, D)
  const float* ctx  = (const float*)d_in[1];  // (N, T, D)
  const float* Wq   = (const float*)d_in[2];
  const float* bq   = (const float*)d_in[3];
  const float* Wk   = (const float*)d_in[4];
  const float* bk   = (const float*)d_in[5];
  const float* Wv   = (const float*)d_in[6];
  const float* bv   = (const float*)d_in[7];
  float* out = (float*)d_out;

  const int Dv = 1024, Xv = 1024, Tv = 4096;
  const int Nv  = in_sizes[0] / (Xv * Dv);    // = 2
  const int Mq  = Nv * Xv;                    // 2048 query rows
  const int Mkv = Nv * Tv;                    // 8192 context rows

  // Workspace: bf16 q (4 MB) + k (16 MB) + v (16 MB) = 36 MB.
  __bf16* qb   = (__bf16*)d_ws;
  __bf16* kb   = qb + (size_t)Mq  * Dv;
  __bf16* vbuf = kb + (size_t)Mkv * Dv;

  dim3 blk(256);
  ca_proj_bf16<<<dim3(Mq / 16, Dv / 512), blk, 0, stream>>>(prev, Wq, bq, qb, Mq, Dv, Dv);
  ca_proj_kv_bf16<<<dim3(Mkv / 16, Dv / 512), blk, 0, stream>>>(ctx, Wk, bk, Wv, bv,
                                                                kb, vbuf, Mkv, Dv, Dv);

  const int waves = Nv * 16 * (Xv / 16);      // N * H * (X/16) = 2048 waves
  ca_flash_attn<<<dim3(waves / 8), blk, 0, stream>>>(qb, kb, vbuf, out);
}
